// NaiveGNN_35244501631341
// MI455X (gfx1250) — matrix operand order repaired and checked by reading, not compiled, not run on hardware
//
#include <hip/hip_runtime.h>
#include <math.h>

// ---------- types for WMMA ----------
typedef _Float16 v16h __attribute__((ext_vector_type(16)));
typedef float    v8f  __attribute__((ext_vector_type(8)));

union F16Frag { v16h v; uint4 q[2]; };

// ---------- ordered-float encoding for uint atomicMax ----------
__device__ __forceinline__ unsigned encf(float x) {
    unsigned u = __float_as_uint(x);
    return (u & 0x80000000u) ? ~u : (u | 0x80000000u);
}
__device__ __forceinline__ float decf(unsigned u) {
    return (u & 0x80000000u) ? __uint_as_float(u & 0x7FFFFFFFu) : __uint_as_float(~u);
}

// ============================================================
// 1) Edge scatter: per pin edge, accumulate sum/max/count in both
//    directions (cell->net and net->cell). One thread per edge.
// ============================================================
__global__ void scatter_kernel(const float* __restrict__ cell_feat,
                               const float* __restrict__ net_feat,
                               const int* __restrict__ src,
                               const int* __restrict__ dst,
                               float* __restrict__ net_sum, unsigned* __restrict__ net_maxe,
                               float* __restrict__ net_cnt,
                               float* __restrict__ cell_sum, unsigned* __restrict__ cell_maxe,
                               float* __restrict__ cell_cnt,
                               int e_pin) {
    int e = blockIdx.x * 256 + threadIdx.x;
    if (e >= e_pin) return;
    int c = src[e], n = dst[e];

    const float4* cf = (const float4*)(cell_feat + (size_t)c * 8);
    float4 a = cf[0], b = cf[1];
    float* ns = net_sum + (size_t)n * 8;
    unsigned* nm = net_maxe + (size_t)n * 8;
    atomicAdd(ns + 0, a.x); atomicAdd(ns + 1, a.y); atomicAdd(ns + 2, a.z); atomicAdd(ns + 3, a.w);
    atomicAdd(ns + 4, b.x); atomicAdd(ns + 5, b.y); atomicAdd(ns + 6, b.z); atomicAdd(ns + 7, b.w);
    atomicMax(nm + 0, encf(a.x)); atomicMax(nm + 1, encf(a.y));
    atomicMax(nm + 2, encf(a.z)); atomicMax(nm + 3, encf(a.w));
    atomicMax(nm + 4, encf(b.x)); atomicMax(nm + 5, encf(b.y));
    atomicMax(nm + 6, encf(b.z)); atomicMax(nm + 7, encf(b.w));
    atomicAdd(net_cnt + n, 1.0f);

    const float4* nf = (const float4*)(net_feat + (size_t)n * 8);
    float4 u = nf[0], v = nf[1];
    float* cs = cell_sum + (size_t)c * 8;
    unsigned* cm = cell_maxe + (size_t)c * 8;
    atomicAdd(cs + 0, u.x); atomicAdd(cs + 1, u.y); atomicAdd(cs + 2, u.z); atomicAdd(cs + 3, u.w);
    atomicAdd(cs + 4, v.x); atomicAdd(cs + 5, v.y); atomicAdd(cs + 6, v.z); atomicAdd(cs + 7, v.w);
    atomicMax(cm + 0, encf(u.x)); atomicMax(cm + 1, encf(u.y));
    atomicMax(cm + 2, encf(u.z)); atomicMax(cm + 3, encf(u.w));
    atomicMax(cm + 4, encf(v.x)); atomicMax(cm + 5, encf(v.y));
    atomicMax(cm + 6, encf(v.z)); atomicMax(cm + 7, encf(v.w));
    atomicAdd(cell_cnt + c, 1.0f);
}

// ============================================================
// 2) Node MLP: h = tanh([feat, mean, max] @ W(24x64) + b).
//    256 threads = 4 nodes x 64 output columns. W cached in LDS.
//    Writes f32 (nets -> hn) and/or f16 (cells -> A matrix col 0..63).
// ============================================================
__global__ void node_mlp_kernel(const float* __restrict__ feat,
                                const float* __restrict__ sum,
                                const unsigned* __restrict__ maxe,
                                const float* __restrict__ cnt,
                                const float* __restrict__ W,
                                const float* __restrict__ b,
                                float* __restrict__ out_f32,
                                _Float16* __restrict__ out_f16,
                                int n, int stride16) {
    __shared__ float sW[24 * 64];
    __shared__ float sx[4][24];
    int tid = threadIdx.x;
    for (int i = tid; i < 24 * 64; i += 256) sW[i] = W[i];
    int g = tid >> 6, t = tid & 63;
    int node = blockIdx.x * 4 + g;
    bool ok = node < n;
    if (ok && t < 24) {
        float v;
        if (t < 8) {
            v = feat[(size_t)node * 8 + t];
        } else {
            float c = cnt[node];
            if (t < 16) v = sum[(size_t)node * 8 + (t - 8)] / fmaxf(c, 1.0f);
            else        v = (c > 0.0f) ? decf(maxe[(size_t)node * 8 + (t - 16)]) : 0.0f;
        }
        sx[g][t] = v;
    }
    __syncthreads();
    if (!ok) return;
    float acc = b[t];
#pragma unroll
    for (int k = 0; k < 24; k++) acc += sx[g][k] * sW[k * 64 + t];
    float h = tanhf(acc);
    if (out_f32) out_f32[(size_t)node * 64 + t] = h;
    if (out_f16) out_f16[(size_t)node * stride16 + t] = (_Float16)h;
}

// ============================================================
// 3) Pin MLP + edge-weighted message scatter.
//    One wave32 per edge: lanes 0..15 compute hp[j]*W_ew[j],
//    shfl-xor reduce -> ew, then each lane atomically adds 2 of the
//    64 message components ew*hn[dst][k] into neigh_sum[src].
// ============================================================
__global__ void pin_msg_kernel(const float* __restrict__ pin_feat,
                               const float* __restrict__ W_pin,
                               const float* __restrict__ b_pin,
                               const float* __restrict__ W_ew,
                               const float* __restrict__ b_ew,
                               const int* __restrict__ src,
                               const int* __restrict__ dst,
                               const float* __restrict__ hn,
                               float* __restrict__ neigh,
                               int e_pin) {
    int lane = threadIdx.x & 31;
    int e = blockIdx.x * 8 + (threadIdx.x >> 5);
    if (e >= e_pin) return;
    float part = 0.0f;
    if (lane < 16) {
        float acc = b_pin[lane];
        const float* pf = pin_feat + (size_t)e * 8;
#pragma unroll
        for (int k = 0; k < 8; k++) acc += pf[k] * W_pin[k * 16 + lane];
        part = tanhf(acc) * W_ew[lane];
    }
#pragma unroll
    for (int off = 16; off > 0; off >>= 1) part += __shfl_xor(part, off, 32);
    float ew = tanhf(part + b_ew[0]);
    int c = src[e], n = dst[e];
    const float* h = hn + (size_t)n * 64;
    float* o = neigh + (size_t)c * 64;
    atomicAdd(o + lane,      h[lane] * ew);
    atomicAdd(o + lane + 32, h[lane + 32] * ew);
}

// ============================================================
// 4) Divide neighbor sums by degree, store f16 into A matrix cols 64..127.
// ============================================================
__global__ void neigh_div_kernel(const float* __restrict__ neigh,
                                 const float* __restrict__ cnt,
                                 _Float16* __restrict__ A, int total) {
    int i = blockIdx.x * 256 + threadIdx.x;
    if (i >= total) return;
    int c = i >> 6, t = i & 63;
    A[(size_t)c * 128 + 64 + t] = (_Float16)(neigh[i] / fmaxf(cnt[c], 1.0f));
}

// ============================================================
// 5) Pack [W_self; W_neigh] (128x64 f32) into f16 pre-swizzled into the
//    per-lane B fragment order used here (contiguous k, as A) for v_wmma_f32_16x16x32_f16:
//    Wf[((kb*4 + ctile)*32 + lane)*16 + i] = Wcat[kb*32 + (lane>>4)*16 + i][ctile*16 + (lane&15)]
// ============================================================
__global__ void packw_kernel(const float* __restrict__ Wself,
                             const float* __restrict__ Wneigh,
                             _Float16* __restrict__ Wf) {
    int idx = blockIdx.x * 256 + threadIdx.x;
    if (idx >= 8192) return;
    int i = idx & 15;
    int lane = (idx >> 4) & 31;
    int t = (idx >> 9) & 3;
    int kb = idx >> 11;
    int row = kb * 32 + (lane >> 4) * 16 + i;
    int col = t * 16 + (lane & 15);
    float v = (row < 64) ? Wself[row * 64 + col] : Wneigh[(row - 64) * 64 + col];
    Wf[idx] = (_Float16)v;
}

// ============================================================
// 6) hidden = A(400k x 128, f16) @ Wcat(128 x 64, f16) + b_sage, f32 out.
//    8 waves/block, each wave owns one 16x16 output tile; block covers
//    32 rows x 64 cols. K=128 in 4 x v_wmma_f32_16x16x32_f16 steps.
//    B fragments come pre-swizzled from LDS (16 KB).
// ============================================================
__global__ __launch_bounds__(256) void hidden_wmma_kernel(
        const _Float16* __restrict__ A,
        const _Float16* __restrict__ Wf,
        const float* __restrict__ b_sage,
        float* __restrict__ hidden, int n_cell) {
    __shared__ alignas(16) _Float16 sW[8192];
    for (int i = threadIdx.x; i < 1024; i += 256)
        ((uint4*)sW)[i] = ((const uint4*)Wf)[i];
    __syncthreads();
    int blk_row = blockIdx.x * 32;
    if (blk_row >= n_cell) return;
    int wave = threadIdx.x >> 5, lane = threadIdx.x & 31;
    int row0 = blk_row + (wave & 1) * 16;
    int ct = wave >> 1;             // column tile 0..3
    int col0 = ct * 16;
    int m = lane & 15, hs = lane >> 4;

    v8f acc = {};
#pragma unroll
    for (int kb = 0; kb < 4; kb++) {
        int k0 = kb * 32;
        F16Frag af, bf;
        // A fragment: lane (m, hs) holds row row0+m, K = {k0+hs*16..+15}
        const _Float16* ap = A + (size_t)(row0 + m) * 128 + k0 + hs * 16;
        af.q[0] = *(const uint4*)ap;
        af.q[1] = *(const uint4*)(ap + 8);
        // B fragment: contiguous 16 halves per lane thanks to pre-swizzle
        const uint4* bp = (const uint4*)(sW + ((kb * 4 + ct) * 32 + lane) * 16);
        bf.q[0] = bp[0];
        bf.q[1] = bp[1];
        acc = __builtin_amdgcn_wmma_f32_16x16x32_f16(
            false, af.v, false, bf.v, (short)0, acc, false, false);
    }
    float bias = b_sage[col0 + m];
#pragma unroll
    for (int v = 0; v < 8; v++)
        hidden[(size_t)(row0 + hs * 8 + v) * 64 + col0 + m] = acc[v] + bias;
}

// ============================================================
// 7) Edge readouts: wave32 per edge; dot(192) and dot(320) via
//    strided lane partials + shfl-xor reduction.
// ============================================================
__global__ void edge_out_kernel(const float* __restrict__ hidden,
                                const float* __restrict__ hn,
                                const float* __restrict__ cell_size,
                                const float* __restrict__ W_dis, const float* __restrict__ b_dis,
                                const float* __restrict__ W_def, const float* __restrict__ b_def,
                                const int* __restrict__ fathers, const int* __restrict__ sons,
                                const int* __restrict__ gfs, const int* __restrict__ fs_nets,
                                const int* __restrict__ gf_nets,
                                float* __restrict__ out_dis, float* __restrict__ out_def,
                                int e2) {
    int lane = threadIdx.x & 31;
    int e = blockIdx.x * 8 + (threadIdx.x >> 5);
    if (e >= e2) return;
    int f = fathers[e], s = sons[e], g = gfs[e], nf = fs_nets[e], ng = gf_nets[e];
    const float* Hf = hidden + (size_t)f * 64;
    const float* Hs = hidden + (size_t)s * 64;
    const float* Hg = hidden + (size_t)g * 64;
    const float* Nf = hn + (size_t)nf * 64;
    const float* Ng = hn + (size_t)ng * 64;
    float d1 = 0.0f, d2 = 0.0f;
#pragma unroll
    for (int i0 = 0; i0 < 64; i0 += 32) {
        int i = i0 + lane;
        float hf = Hf[i], hsv = Hs[i], hg = Hg[i], vnf = Nf[i], vng = Ng[i];
        d1 += hf * W_dis[i] + hsv * W_dis[64 + i] + vnf * W_dis[128 + i];
        d2 += hg * W_def[i] + hf * W_def[64 + i] + hsv * W_def[128 + i]
            + vng * W_def[192 + i] + vnf * W_def[256 + i];
    }
#pragma unroll
    for (int off = 16; off > 0; off >>= 1) {
        d1 += __shfl_xor(d1, off, 32);
        d2 += __shfl_xor(d2, off, 32);
    }
    if (lane == 0) {
        float dis = expf(-2.0f + 15.0f * tanhf(d1 + b_dis[0]));
        const float* cf = cell_size + (size_t)f * 2;
        const float* cs = cell_size + (size_t)s * 2;
        float bx = (cf[0] + cs[0]) * 0.5f;
        float by = (cf[1] + cs[1]) * 0.5f;
        out_dis[e] = dis + fminf(bx, by);
        out_def[e] = tanhf(d2 + b_def[0]) * 6.283185307179586f;
    }
}

// ============================================================
// launch
// ============================================================
extern "C" void kernel_launch(void* const* d_in, const int* in_sizes, int n_in,
                              void* d_out, int out_size, void* d_ws, size_t ws_size,
                              hipStream_t stream) {
    const float* cell_feat = (const float*)d_in[0];
    const float* net_feat  = (const float*)d_in[1];
    const float* pin_feat  = (const float*)d_in[2];
    const float* cell_size = (const float*)d_in[3];
    const float* W_cell = (const float*)d_in[4];
    const float* b_cell = (const float*)d_in[5];
    const float* W_net  = (const float*)d_in[6];
    const float* b_net  = (const float*)d_in[7];
    const float* W_pin  = (const float*)d_in[8];
    const float* b_pin  = (const float*)d_in[9];
    const float* W_ew   = (const float*)d_in[10];
    const float* b_ew   = (const float*)d_in[11];
    const float* W_self = (const float*)d_in[12];
    const float* W_neigh= (const float*)d_in[13];
    const float* b_sage = (const float*)d_in[14];
    const float* W_dis  = (const float*)d_in[15];
    const float* b_dis  = (const float*)d_in[16];
    const float* W_def  = (const float*)d_in[17];
    const float* b_def  = (const float*)d_in[18];
    const int* pins_src = (const int*)d_in[19];
    const int* pins_dst = (const int*)d_in[20];
    const int* fathers  = (const int*)d_in[21];
    const int* sons     = (const int*)d_in[22];
    const int* gfs      = (const int*)d_in[23];
    const int* fs_nets  = (const int*)d_in[24];
    const int* gf_nets  = (const int*)d_in[25];

    const int n_cell = in_sizes[0] / 8;
    const int n_net  = in_sizes[1] / 8;
    const int e_pin  = in_sizes[19];
    const int e2     = in_sizes[21];

    // carve workspace; zero-initialized accumulators first (one memset)
    char* p = (char*)d_ws;
    auto carve = [&](size_t bytes) -> void* {
        void* r = (void*)p;
        p += (bytes + 255) & ~(size_t)255;
        return r;
    };
    float*    net_sum   = (float*)   carve((size_t)n_net * 8 * 4);
    unsigned* net_maxe  = (unsigned*)carve((size_t)n_net * 8 * 4);
    float*    net_cnt   = (float*)   carve((size_t)n_net * 4);
    float*    cell_sum  = (float*)   carve((size_t)n_cell * 8 * 4);
    unsigned* cell_maxe = (unsigned*)carve((size_t)n_cell * 8 * 4);
    float*    cell_cnt  = (float*)   carve((size_t)n_cell * 4);
    float*    neigh     = (float*)   carve((size_t)n_cell * 64 * 4);
    size_t zero_bytes = (size_t)(p - (char*)d_ws);
    float*    hn     = (float*)   carve((size_t)n_net * 64 * 4);
    _Float16* A      = (_Float16*)carve((size_t)n_cell * 128 * 2);
    _Float16* Wf     = (_Float16*)carve((size_t)8192 * 2);
    float*    hidden = (float*)   carve((size_t)n_cell * 64 * 4);

    float* out_dis = (float*)d_out;
    float* out_def = out_dis + e2;

    hipMemsetAsync(d_ws, 0, zero_bytes, stream);

    // 1) segment sum/max/count both directions
    scatter_kernel<<<(e_pin + 255) / 256, 256, 0, stream>>>(
        cell_feat, net_feat, pins_src, pins_dst,
        net_sum, net_maxe, net_cnt, cell_sum, cell_maxe, cell_cnt, e_pin);

    // 2) node MLPs (cells -> A f16 cols 0..63; nets -> hn f32)
    node_mlp_kernel<<<(n_cell + 3) / 4, 256, 0, stream>>>(
        cell_feat, cell_sum, cell_maxe, cell_cnt, W_cell, b_cell,
        (float*)nullptr, A, n_cell, 128);
    node_mlp_kernel<<<(n_net + 3) / 4, 256, 0, stream>>>(
        net_feat, net_sum, net_maxe, net_cnt, W_net, b_net,
        hn, (_Float16*)nullptr, n_net, 0);

    // 3) pin MLP + weighted message scatter
    pin_msg_kernel<<<(e_pin + 7) / 8, 256, 0, stream>>>(
        pin_feat, W_pin, b_pin, W_ew, b_ew, pins_src, pins_dst, hn, neigh, e_pin);

    // 4) mean-normalize neighbor messages into A f16 cols 64..127
    neigh_div_kernel<<<((n_cell * 64) + 255) / 256, 256, 0, stream>>>(
        neigh, cell_cnt, A, n_cell * 64);

    // 5) pack fused [W_self; W_neigh] into WMMA B-fragment layout
    packw_kernel<<<32, 256, 0, stream>>>(W_self, W_neigh, Wf);

    // 6) WMMA GEMM: hidden = A @ Wcat + b_sage
    hidden_wmma_kernel<<<(n_cell + 31) / 32, 256, 0, stream>>>(
        A, Wf, b_sage, hidden, n_cell);

    // 7) edge readouts
    edge_out_kernel<<<(e2 + 7) / 8, 256, 0, stream>>>(
        hidden, hn, cell_size, W_dis, b_dis, W_def, b_def,
        fathers, sons, gfs, fs_nets, gf_nets, out_dis, out_def, e2);
}
